// KANLinear_20615843021208
// MI455X (gfx1250) — matrix operand order repaired and checked by reading, not compiled, not run on hardware
//
#include <hip/hip_runtime.h>

typedef __attribute__((ext_vector_type(16))) _Float16 v16h;
typedef __attribute__((ext_vector_type(8)))  _Float16 v8h;
typedef __attribute__((ext_vector_type(8)))  float    v8f;
typedef __attribute__((ext_vector_type(4)))  unsigned int u32x4;
typedef __attribute__((ext_vector_type(8)))  int      i32x8;
typedef __attribute__((ext_vector_type(4)))  int      i32x4;

#define NROWS 32768
#define INF   256
#define OUTF  256
#define KTOT  2304   // 256 (base) + 256*8 (spline)
#define NKC   72     // KTOT / 32
#define BM    64     // rows per block
#define APITCH 40    // A-chunk LDS row pitch in halves (32 + 8 pad) = 80B
#define ABUF  (BM * APITCH)
#define XPITCH 260   // x-tile LDS row pitch in floats (256 + 4 pad) = 1040B

// ---------------------------------------------------------------------------
// Pack [base_weight | spline_weight] (f32) into f16 in WMMA-B-fragment-linear
// order: addr = (((kc*16 + ct)*32 + lane)*16 + e) halves, where
//   Kloc = (e<8 ? 0 : 16) + (lane>=16 ? 8 : 0) + (e & 7)   (B 32x16 f16 layout)
//   N    = ct*16 + (lane & 15)
// ---------------------------------------------------------------------------
__global__ __launch_bounds__(256) void pack_weights_kernel(
    const float* __restrict__ bw,      // [OUTF][INF]
    const float* __restrict__ sw,      // [OUTF][INF][8]
    _Float16* __restrict__ wf)         // [KTOT*OUTF] halves, fragment-linear
{
    int idx  = blockIdx.x * 256 + threadIdx.x;
    int e    = idx & 15;
    int lane = (idx >> 4) & 31;
    int ct   = (idx >> 9) & 15;
    int kc   = idx >> 13;

    int Kloc = ((e >> 3) << 4) + ((lane >> 4) << 3) + (e & 7);
    int k    = kc * 32 + Kloc;
    int o    = ct * 16 + (lane & 15);

    float v;
    if (k < INF) {
        v = bw[o * INF + k];
    } else {
        int kp = k - INF;                       // i = kp/8, s = kp%8
        v = sw[(o * INF + (kp >> 3)) * 8 + (kp & 7)];
    }
    wf[idx] = (_Float16)v;
}

// ---------------------------------------------------------------------------
// Generate one 64x32 f16 A-chunk row-slice for this thread into LDS,
// reading x from the LDS-resident x-tile.
// ---------------------------------------------------------------------------
__device__ __forceinline__ void gen_chunk(int kc, _Float16* dst,
                                          const float* xrowLds, float pw, int gq)
{
    if (kc < 8) {
        // base region: prelu(x[n][kc*32 + gq*8 .. +8])
        const float* xr = xrowLds + kc * 32 + gq * 8;
        float4 p0 = ((const float4*)xr)[0];
        float4 p1 = ((const float4*)xr)[1];
        float vals[8] = {p0.x, p0.y, p0.z, p0.w, p1.x, p1.y, p1.z, p1.w};
        v8h o;
#pragma unroll
        for (int e = 0; e < 8; ++e) {
            float v = vals[e];
            v = (v >= 0.0f) ? v : pw * v;
            o[e] = (_Float16)v;
        }
        *(v8h*)dst = o;
    } else {
        // spline region: 8 cardinal cubic B-spline bases of x[n][feat]
        // knots t_j = -1 + (j-3)*0.4 ; cell m: t_m <= x < t_{m+1}
        int   feat = (kc - 8) * 4 + gq;          // 0..255
        float xv   = xrowLds[feat];
        float tpos = (xv + 2.2f) * 2.5f;         // (x - t_0)/h
        float fm   = floorf(tpos);
        int   m    = (int)fm;
        float u    = tpos - fm;
        bool  inr  = (tpos >= 0.0f) && (m <= 10);
        float um = 1.0f - u;
        float u2 = u * u, u3 = u2 * u;
        float w0 = um * um * um * (1.0f / 6.0f);
        float w1 = (3.0f * u3 - 6.0f * u2 + 4.0f) * (1.0f / 6.0f);
        float w2 = (-3.0f * u3 + 3.0f * u2 + 3.0f * u + 1.0f) * (1.0f / 6.0f);
        float w3 = u3 * (1.0f / 6.0f);
        v8h o;
#pragma unroll
        for (int j = 0; j < 8; ++j) {
            int d = j - m + 3;                   // basis j nonzero iff d in [0,3]
            float wv = (d == 0) ? w0
                     : (d == 1) ? w1
                     : (d == 2) ? w2
                     : (d == 3) ? w3 : 0.0f;
            o[j] = (_Float16)(inr ? wv : 0.0f);
        }
        *(v8h*)dst = o;
    }
}

// ---------------------------------------------------------------------------
// Fused KAN forward: out = prelu(x) @ Wb^T + Bspline(x) . Ws as one
// M=32768, N=256, K=2304 f16 WMMA GEMM.
//  - x tile (64x256 f32) staged LDS-resident once via TDM tensor_load_to_lds
//    (row-padded by the TDM to a 260-dword pitch for bank-conflict-free reads)
//  - A chunks generated on the fly, double-buffered, 1 barrier per K-chunk
//  - B fragments pre-packed fragment-linear in L2, 32B/lane coalesced
//  - wave tile 4 row-blocks x 2 col-tiles: zero B duplication across waves
// ---------------------------------------------------------------------------
__global__ __launch_bounds__(256) void kan_wmma_kernel(
    const float* __restrict__ x,        // [NROWS][INF]
    const _Float16* __restrict__ wf,    // packed weights (fragment-linear)
    const float* __restrict__ prelu_w,  // [1]
    float* __restrict__ out)            // [NROWS][OUTF]
{
    __shared__ float     smX[BM * XPITCH];   // 65 KB x-tile (padded rows)
    __shared__ _Float16  smA[2 * ABUF];      // 10 KB double-buffered A-chunks

    const int t    = threadIdx.x;
    const int lane = t & 31;
    const int wave = t >> 5;                 // 0..7
    const int row0 = blockIdx.x * BM;

    const float pw = prelu_w[0];

    // ---- stage x tile into LDS via Tensor Data Mover --------------------
    if (wave == 0) {
        unsigned long long ga = (unsigned long long)(const void*)(x + (size_t)row0 * INF);
        unsigned lds = (unsigned)(unsigned long long)(const void*)&smX[0];

        u32x4 g0;
        g0[0] = 1u;                                   // count=1, user descriptor
        g0[1] = lds;                                  // lds_addr
        g0[2] = (unsigned)(ga & 0xFFFFFFFFu);         // global_addr[31:0]
        g0[3] = (unsigned)((ga >> 32) & 0x1FFFFFFu)   // global_addr[56:32]
              | 0x80000000u;                          // type=2 ("image")

        i32x8 g1;
        g1[0] = (2 << 16)                             // data_size = 4B
              | (1 << 20)                             // pad_enable
              | (7 << 22)                             // pad_interval: 256 dwords
              | (3 << 25);                            // pad_amount: 4 dwords
        g1[1] = (INF & 0xFFFF) << 16;                 // tensor_dim0[15:0] = 256
        g1[2] = (INF >> 16) | ((BM & 0xFFFF) << 16);  // dim0 hi | tensor_dim1 lo = 64
        g1[3] = (BM >> 16) | ((INF & 0xFFFF) << 16);  // dim1 hi | tile_dim0 = 256
        g1[4] = BM;                                   // tile_dim1 = 64, tile_dim2 = 0
        g1[5] = INF;                                  // tensor_dim0_stride lo = 256
        g1[6] = 0;                                    // stride0 hi | stride1 lo
        g1[7] = 0;                                    // stride1 hi

        i32x4 gz4 = {0, 0, 0, 0};
        i32x8 gz8 = {0, 0, 0, 0, 0, 0, 0, 0};
        __builtin_amdgcn_tensor_load_to_lds(g0, g1, gz4, gz4, gz8, 0);
        __builtin_amdgcn_s_wait_tensorcnt(0);
    }
    __syncthreads();

    // A-generation mapping: thread -> (row gr, quarter gq)
    const int gr = t >> 2;                   // 0..63
    const int gq = t & 3;                    // 0..3
    const float* xrowLds = smX + gr * XPITCH;
    _Float16* genSlot    = smA + gr * APITCH + gq * 8;

    // WMMA mapping: wave -> 2 col tiles, all 4 row blocks
    const int ct0  = wave * 2;               // col tiles {ct0, ct0+1}
    const int kA   = (lane >> 4) * 8;        // A 16x32 f16 layout half-select
    const int aCol = (lane & 15) * APITCH;

    v8f acc[8] = {};                         // acc[rb*2 + c]

    const v16h* __restrict__ wfrag = (const v16h*)wf;

    // prologue: generate chunk 0 into buffer 0
    gen_chunk(0, genSlot, xrowLds, pw, gq);
    __syncthreads();

    for (int kc = 0; kc < NKC; ++kc) {
        const int buf = (kc & 1) * ABUF;

        // --- B fragments for this chunk (2 x v16h, 32B/lane coalesced) ---
        const v16h* bp = wfrag + ((size_t)kc * 16 + ct0) * 32 + lane;
        v16h b0 = bp[0];
        v16h b1 = bp[32];

        // --- prefetch B region two chunks ahead (global_prefetch_b8) ---
        if (kc + 2 < NKC) {
            const char* pf = (const char*)(wfrag + ((size_t)(kc + 2) * 16 + ct0) * 32)
                             + lane * 64;
            __builtin_prefetch(pf, 0, 1);
        }

        // --- generate next A-chunk into the other buffer (overlaps loads) ---
        if (kc + 1 < NKC)
            gen_chunk(kc + 1, genSlot + (ABUF - buf), xrowLds, pw, gq);

        // --- A fragments: 4 row blocks, 2 x 16B LDS reads each ---
        union { v16h v; v8h h[2]; } a[4];
#pragma unroll
        for (int rb = 0; rb < 4; ++rb) {
            const _Float16* ap = smA + buf + rb * 16 * APITCH + aCol + kA;
            a[rb].h[0] = *(const v8h*)ap;
            a[rb].h[1] = *(const v8h*)(ap + 16);
        }

        // --- 8 WMMAs: single wait, back-to-back issue ---
#pragma unroll
        for (int rb = 0; rb < 4; ++rb) {
            acc[rb * 2 + 0] = __builtin_amdgcn_wmma_f32_16x16x32_f16(
                false, a[rb].v, false, b0, (short)0, acc[rb * 2 + 0], false, false);
            acc[rb * 2 + 1] = __builtin_amdgcn_wmma_f32_16x16x32_f16(
                false, a[rb].v, false, b1, (short)0, acc[rb * 2 + 1], false, false);
        }

        __syncthreads();   // next-chunk generation complete; old buffer free
    }

    // ---- store C (16x16 f32 layout: VGPR g -> M = g or g+8 by lane half)
    const int mOff = (lane >> 4) * 8;
    const int n0   = lane & 15;
#pragma unroll
    for (int rb = 0; rb < 4; ++rb) {
        const int rowOut = row0 + rb * 16 + mOff;
#pragma unroll
        for (int c = 0; c < 2; ++c) {
            const int col = (ct0 + c) * 16 + n0;
#pragma unroll
            for (int g = 0; g < 8; ++g) {
                out[(size_t)(rowOut + g) * OUTF + col] = acc[rb * 2 + c][g];
            }
        }
    }
}

extern "C" void kernel_launch(void* const* d_in, const int* in_sizes, int n_in,
                              void* d_out, int out_size, void* d_ws, size_t ws_size,
                              hipStream_t stream) {
    (void)in_sizes; (void)n_in; (void)out_size; (void)ws_size;

    const float* x   = (const float*)d_in[0];
    // d_in[1] = grid: uniform knots reproduced analytically in-kernel
    const float* bw  = (const float*)d_in[2];
    const float* sw  = (const float*)d_in[3];
    const float* pwv = (const float*)d_in[4];

    _Float16* wf = (_Float16*)d_ws;   // 2304*256*2 B = 1.18 MB

    pack_weights_kernel<<<(KTOT * OUTF) / 256, 256, 0, stream>>>(bw, sw, wf);
    kan_wmma_kernel<<<NROWS / BM, 256, 0, stream>>>(x, wf, pwv, (float*)d_out);
}
